// DigitConvolutionalModel_85813446574657
// MI455X (gfx1250) — compile-verified, matrix operand order checked by reading, not run on hardware
//
#include <hip/hip_runtime.h>

typedef __attribute__((ext_vector_type(16))) _Float16     v16h;
typedef __attribute__((ext_vector_type(8)))  float        v8f;
typedef __attribute__((ext_vector_type(4)))  float        f32x4;
typedef __attribute__((ext_vector_type(2)))  float        f32x2;
typedef __attribute__((ext_vector_type(4)))  unsigned int u32x4;
typedef int v4i __attribute__((vector_size(16)));   // matches builtin param type

#define HID    128
#define OUTN   10
#define KREAL  784
#define KPAD   800
#define ROWS   128          // rows of X per block
#define XSTR   36           // X LDS stride in f32 (144B: 16B-aligned, conflict-free)
#define WSTR   40           // W LDS stride in f16 (80B: 16B-aligned, conflict-free)
#define HSTR   130          // h-tile stride in halves

// ---- gfx1250 async global->LDS path (guarded; fallback = register staging) ----
#if defined(__has_builtin)
#  if __has_builtin(__builtin_amdgcn_global_load_async_to_lds_b128)
#    define USE_ASYNC 1
#  endif
#endif
#ifndef USE_ASYNC
#  define USE_ASYNC 0
#endif

#if USE_ASYNC
#  define ASYNC_B128(l, g, off)                                        \
     __builtin_amdgcn_global_load_async_to_lds_b128(                   \
         (__attribute__((address_space(1))) v4i*)(g),                  \
         (__attribute__((address_space(3))) v4i*)(l), (off), 0)
#  if __has_builtin(__builtin_amdgcn_s_wait_asynccnt)
#    define WAIT_ASYNC() __builtin_amdgcn_s_wait_asynccnt(0)
#  else
#    define WAIT_ASYNC() asm volatile("s_wait_asynccnt 0x0" ::: "memory")
#  endif
#endif

union AF { v16h h; unsigned int u[8]; };

static __device__ __forceinline__ unsigned int pack2(float a, float b) {
    union { _Float16 h[2]; unsigned int u; } p;
    p.h[0] = (_Float16)a; p.h[1] = (_Float16)b;
    return p.u;
}

// ---------------------------------------------------------------------------
// Prologue 1: fold conv into W_eff [128][800] f16 (cols 784..799 zero-padded)
// ---------------------------------------------------------------------------
__global__ void build_weff(const float* __restrict__ w1,
                           const float* __restrict__ cw,
                           _Float16* __restrict__ weff) {
    int idx = blockIdx.x * 256 + threadIdx.x;
    if (idx >= HID * KPAD) return;
    int n = idx / KPAD, k = idx % KPAD;
    float s = 0.0f;
    if (k < KREAL) {
        int r = k / 28, c = k % 28;
        #pragma unroll
        for (int dy = 0; dy < 3; ++dy) {
            #pragma unroll
            for (int dx = 0; dx < 3; ++dx) {
                int rr = r - dy, cc = c - dx;
                if (rr >= 0 && rr < 26 && cc >= 0 && cc < 26)
                    s += cw[dy * 3 + dx] * w1[n * 676 + rr * 26 + cc];
            }
        }
    }
    weff[idx] = (_Float16)s;
}

// ---------------------------------------------------------------------------
// Prologue 2: pad w2 [10][128] -> f16 [16][128] (rows 10..15 zero)
// ---------------------------------------------------------------------------
__global__ void build_w2p(const float* __restrict__ w2,
                          _Float16* __restrict__ w2p) {
    int idx = blockIdx.x * 256 + threadIdx.x;
    if (idx >= 16 * HID) return;
    int n = idx / HID, k = idx % HID;
    float v = (n < OUTN) ? w2[n * HID + k] : 0.0f;
    w2p[idx] = (_Float16)v;
}

// ---------------------------------------------------------------------------
// Fused GEMM: out = relu(X @ Weff^T + b1) @ w2^T + b2
// block = 256 threads (8 waves), 128 rows of X per block
// double-buffered LDS, async global->LDS staging when available
// ---------------------------------------------------------------------------
struct StageBufs {
    float    xs[2][ROWS][XSTR];   // X chunk as f32
    _Float16 wz[2][HID][WSTR];    // Weff chunk as f16
};
union SMem {
    StageBufs a;                  // 36864 + 20480 = 57344 B
    _Float16  hs[8][16][HSTR];    // 33280 B (used only after the K loop)
};

__global__ __launch_bounds__(256, 1)
void fused_mlp(const float* __restrict__ x,
               const _Float16* __restrict__ weff,
               const _Float16* __restrict__ w2p,
               const float* __restrict__ b1,
               const float* __restrict__ b2,
               float* __restrict__ out) {
    __shared__ SMem sm;

    const int t    = threadIdx.x;
    const int lane = t & 31;
    const int wave = t >> 5;
    const int row0 = blockIdx.x * ROWS;

    const int lrow = lane & 15;               // M (A) or N (B) within a tile
    const int khi  = (lane >= 16) ? 8 : 0;    // K-half select (also M-offset in D)

    const int r_loc = t >> 1;                 // cooperative-load row (0..127)
    const int half  = t & 1;                  // which 16-wide half-chunk

    v8f acc[8] = {};

#if !USE_ASYNC
    f32x4 px[4]; u32x4 pw[2]; bool pxv = false;
#endif

    // ---- stage-issue: start global traffic for chunk kc into buffer `buf` ----
    auto stage_issue = [&](int buf, int kc) {
        const int kbase = kc + half * 16;
#if USE_ASYNC
        if (kbase < KREAL) {
            const float* g = x + (size_t)(row0 + r_loc) * KREAL + kbase;
            float*       l = &sm.a.xs[buf][r_loc][half * 16];
            ASYNC_B128(l, g, 0);  ASYNC_B128(l, g, 16);
            ASYNC_B128(l, g, 32); ASYNC_B128(l, g, 48);
        } else {
            f32x4 z = {};
            f32x4* d = (f32x4*)&sm.a.xs[buf][r_loc][half * 16];
            d[0] = z; d[1] = z; d[2] = z; d[3] = z;
        }
        const _Float16* gw = weff + (size_t)r_loc * KPAD + kc + half * 16;
        _Float16*       lw = &sm.a.wz[buf][r_loc][half * 16];
        ASYNC_B128(lw, gw, 0); ASYNC_B128(lw, gw, 16);
#else
        pxv = kbase < KREAL;
        if (pxv) {
            const f32x4* s = (const f32x4*)(x + (size_t)(row0 + r_loc) * KREAL + kbase);
            px[0] = __builtin_nontemporal_load(s + 0);
            px[1] = __builtin_nontemporal_load(s + 1);
            px[2] = __builtin_nontemporal_load(s + 2);
            px[3] = __builtin_nontemporal_load(s + 3);
        }
        const u32x4* ws = (const u32x4*)(weff + (size_t)r_loc * KPAD + kc + half * 16);
        pw[0] = ws[0]; pw[1] = ws[1];
#endif
    };

    // ---- stage-complete: make buffer `buf` valid (before the barrier) ----
    auto stage_complete = [&](int buf) {
#if USE_ASYNC
        (void)buf;
        WAIT_ASYNC();
#else
        f32x4* d = (f32x4*)&sm.a.xs[buf][r_loc][half * 16];
        if (pxv) { d[0] = px[0]; d[1] = px[1]; d[2] = px[2]; d[3] = px[3]; }
        else     { f32x4 z = {}; d[0] = z; d[1] = z; d[2] = z; d[3] = z; }
        u32x4* dw = (u32x4*)&sm.a.wz[buf][r_loc][half * 16];
        dw[0] = pw[0]; dw[1] = pw[1];
#endif
    };

    // ---- compute one K-chunk from buffer `buf` ----
    auto compute = [&](int buf) {
        AF a;
        const float* abase = &sm.a.xs[buf][wave * 16 + lrow][0];
        #pragma unroll
        for (int v = 0; v < 8; ++v) {
            int k = ((v >= 4) ? 16 : 0) + khi + 2 * (v & 3);
            f32x2 two = *(const f32x2*)(abase + k);        // ds_load_b64
            a.u[v] = pack2(two.x, two.y);                  // f32 -> packed f16
        }
        #pragma unroll
        for (int nt = 0; nt < 8; ++nt) {
            AF b;
            const _Float16* bbase = &sm.a.wz[buf][nt * 16 + lrow][0];
            #pragma unroll
            for (int v = 0; v < 8; ++v) {
                int k = ((v >= 4) ? 16 : 0) + khi + 2 * (v & 3);
                b.u[v] = *(const unsigned int*)(bbase + k);
            }
            acc[nt] = __builtin_amdgcn_wmma_f32_16x16x32_f16(
                false, a.h, false, b.h, (short)0, acc[nt], false, false);
        }
    };

    // ---- software pipeline over 25 K-chunks of 32 ----
    stage_issue(0, 0);
    stage_complete(0);
    __syncthreads();
    int cur = 0;
    for (int kc = 0; kc < KPAD; kc += 32) {
        const bool more = (kc + 32) < KPAD;
        if (more) stage_issue(cur ^ 1, kc + 32);
        compute(cur);
        if (more) stage_complete(cur ^ 1);
        __syncthreads();                 // publishes next buffer / frees hs space
        cur ^= 1;
    }

    // ---- bias + ReLU, write h tile (f16) to wave-private LDS ----
    #pragma unroll
    for (int nt = 0; nt < 8; ++nt) {
        float bv = b1[nt * 16 + lrow];
        #pragma unroll
        for (int r = 0; r < 8; ++r) {
            float hv = acc[nt][r] + bv;
            hv = hv > 0.0f ? hv : 0.0f;
            sm.hs[wave][r + khi][nt * 16 + lrow] = (_Float16)hv;   // m = r + khi
        }
    }

    // ---- second GEMM: h [16x128] @ w2p^T [128x16] (K = 4 chunks of 32) ----
    v8f acc2 = {};
    #pragma unroll
    for (int kc2 = 0; kc2 < HID; kc2 += 32) {
        AF a2, bf;
        const _Float16* abase = &sm.hs[wave][lrow][0];
        const _Float16* bbase = w2p + lrow * HID;
        #pragma unroll
        for (int v = 0; v < 8; ++v) {
            int k = kc2 + ((v >= 4) ? 16 : 0) + khi + 2 * (v & 3);
            a2.u[v] = *(const unsigned int*)(abase + k);
            bf.u[v] = *(const unsigned int*)(bbase + k);
        }
        acc2 = __builtin_amdgcn_wmma_f32_16x16x32_f16(
            false, a2.h, false, bf.h, (short)0, acc2, false, false);
    }

    // ---- epilogue: +b2, store [rows x 10] ----
    if (lrow < OUTN) {
        float b2v = b2[lrow];
        #pragma unroll
        for (int r = 0; r < 8; ++r) {
            int row = row0 + wave * 16 + r + khi;   // m = r + khi
            out[(size_t)row * OUTN + lrow] = acc2[r] + b2v;
        }
    }
}

// ---------------------------------------------------------------------------
// Host launcher
// ---------------------------------------------------------------------------
extern "C" void kernel_launch(void* const* d_in, const int* in_sizes, int n_in,
                              void* d_out, int out_size, void* d_ws, size_t ws_size,
                              hipStream_t stream) {
    const float* x      = (const float*)d_in[0];
    const float* conv_w = (const float*)d_in[1];
    const float* w1     = (const float*)d_in[2];
    const float* b1     = (const float*)d_in[3];
    const float* w2     = (const float*)d_in[4];
    const float* b2     = (const float*)d_in[5];
    float*       out    = (float*)d_out;

    _Float16* weff = (_Float16*)d_ws;                                   // 200 KB
    _Float16* w2p  = (_Float16*)((char*)d_ws + (size_t)HID * KPAD * 2); //   4 KB

    const int B = in_sizes[0] / KREAL;   // 65536

    build_weff<<<(HID * KPAD + 255) / 256, 256, 0, stream>>>(w1, conv_w, weff);
    build_w2p <<<(16 * HID + 255) / 256, 256, 0, stream>>>(w2, w2p);
    fused_mlp <<<B / ROWS, 256, 0, stream>>>(x, weff, w2p, b1, b2, out);
}
